// Decoder_81819126989051
// MI455X (gfx1250) — matrix operand order repaired and checked by reading, not compiled, not run on hardware
//
#include <hip/hip_runtime.h>
#include <math.h>

typedef _Float16 half_t;
typedef __attribute__((ext_vector_type(8)))  _Float16 v8h;
typedef __attribute__((ext_vector_type(16))) _Float16 v16h;
typedef __attribute__((ext_vector_type(8)))  float    v8f;

#define BLOCK 256                 // 8 waves (wave32) -> one workgroup per WGP
#define WPB   8                   // waves per block
#define AS    264                 // panel row stride in halfs (256 + 8 pad -> uniform 2-way LDS banking)

// LDS layout (halfs), dynamic smem starts at LDS offset 0 (no static __shared__)
#define W1_OFF 0                  // W1^T resident: 256 rows x AS
#define W2_OFF (256 * AS)         // W2^T resident: 128 rows x AS
#define AP_OFF (W2_OFF + 128 * AS)// 8 per-wave 16 x AS activation panels
#define LDS_HALFS (AP_OFF + WPB * 16 * AS)
#define LDS_BYTES (LDS_HALFS * 2) // 270336 B <= 320 KB WGP LDS

// ---- weight prep: f32 [in][out] -> f16 [out][in] (transposed) ----
__global__ __launch_bounds__(256) void decoder_prep(
    const float* __restrict__ W1, const float* __restrict__ W2,
    half_t* __restrict__ W1T, half_t* __restrict__ W2T)
{
  int id = blockIdx.x * 256 + threadIdx.x;      // 0 .. 65535
  {
    int n = id >> 8, k = id & 255;              // n<256, k<256
    W1T[n * 256 + k] = (half_t)W1[k * 256 + n]; // W1 is [256][256]
  }
  if (id < 128 * 256) {
    int n = id >> 8, k = id & 255;              // n<128, k<256
    W2T[n * 256 + k] = (half_t)W2[k * 128 + n]; // W2 is [256][128]
  }
}

__global__ __launch_bounds__(BLOCK) void decoder_mlp(
    const float* __restrict__ z, const long long* __restrict__ ei,
    const half_t* __restrict__ W1T, const float* __restrict__ b1,
    const half_t* __restrict__ W2T, const float* __restrict__ b2,
    const float* __restrict__ W3,  const float* __restrict__ b3,
    float* __restrict__ out, int E)
{
  extern __shared__ half_t smem[];

  const int tid  = threadIdx.x;
  const int wave = tid >> 5;
  const int lane = tid & 31;
  const int hi   = lane >> 4;                 // 0 | 1 (half-wave)
  const int l15  = lane & 15;

  // ---- one-time: stream both weight matrices into LDS via CDNA5 async copy ----
  // GV mode: 64-bit global address in VADDR, LDS byte address in VDST, SADDR=off.
#pragma unroll 1
  for (int it = 0; it < 32; ++it) {           // W1^T: 8192 x 16B chunks / 256 thr
    int id = it * BLOCK + tid;
    int r = id >> 5, s = id & 31;
    unsigned la = (unsigned)((W1_OFF + r * AS + s * 8) * 2);
    unsigned long long ga = (unsigned long long)(const void*)&W1T[r * 256 + s * 8];
    asm volatile("global_load_async_to_lds_b128 %0, %1, off" :: "v"(la), "v"(ga) : "memory");
  }
#pragma unroll 1
  for (int it = 0; it < 16; ++it) {           // W2^T: 4096 x 16B chunks / 256 thr
    int id = it * BLOCK + tid;
    int r = id >> 5, s = id & 31;
    unsigned la = (unsigned)((W2_OFF + r * AS + s * 8) * 2);
    unsigned long long ga = (unsigned long long)(const void*)&W2T[r * 256 + s * 8];
    asm volatile("global_load_async_to_lds_b128 %0, %1, off" :: "v"(la), "v"(ga) : "memory");
  }
  asm volatile("s_wait_asynccnt 0x0" ::: "memory");
  __syncthreads();                            // only barrier in the kernel

  half_t* A = &smem[AP_OFF + wave * 16 * AS]; // this wave's private panel
  const half_t* W1L = &smem[W1_OFF];
  const half_t* W2L = &smem[W2_OFF];

  const int nT = (E + 15) >> 4;               // 16-edge tiles
  const int stride = (int)gridDim.x * WPB;

  for (int t = blockIdx.x * WPB + wave; t < nT; t += stride) {
    {                                         // prefetch next tile's index lines
      int tn = t + stride;
      if (tn < nT) {
        __builtin_prefetch(&ei[(size_t)tn * 16], 0, 0);
        __builtin_prefetch(&ei[(size_t)E + (size_t)tn * 16], 0, 0);
      }
    }
    const int ebase = t * 16;

    // ---------- gather: row m = concat(z[src_m], z[dst_m]) as f16 ----------
    for (int m = 0; m < 16; ++m) {
      int e = ebase + m;                      // wave-uniform
      long long es = 0, ed = 0;
      if (e < E) { es = ei[e]; ed = ei[(size_t)E + e]; }
      float4 vs = ((const float4*)(z + (size_t)es * 128))[lane];
      float4 vd = ((const float4*)(z + (size_t)ed * 128))[lane];
      union { half_t h[4]; unsigned long long u; } ps, pd;
      ps.h[0] = (half_t)vs.x; ps.h[1] = (half_t)vs.y;
      ps.h[2] = (half_t)vs.z; ps.h[3] = (half_t)vs.w;
      pd.h[0] = (half_t)vd.x; pd.h[1] = (half_t)vd.y;
      pd.h[2] = (half_t)vd.z; pd.h[3] = (half_t)vd.w;
      *(unsigned long long*)&A[m * AS + lane * 4]       = ps.u;
      *(unsigned long long*)&A[m * AS + 128 + lane * 4] = pd.u;
    }

    // ---------- layer 1: [16x256] @ [256x256] ----------
    v8f acc1[16];
#pragma unroll
    for (int n = 0; n < 16; ++n) acc1[n] = (v8f){};

#pragma unroll
    for (int kc = 0; kc < 8; ++kc) {
      // A fragment: lane<16 -> K 0-15 ; lane>=16 -> K 16-31
      int ar = l15 * AS + kc * 32 + hi * 16;
      v8h alo = *(const v8h*)&A[ar];
      v8h ahi = *(const v8h*)&A[ar + 8];
      v16h a = __builtin_shufflevector(alo, ahi, 0,1,2,3,4,5,6,7,8,9,10,11,12,13,14,15);
#pragma unroll
      for (int n = 0; n < 16; ++n) {
        int br = (n * 16 + l15) * AS + kc * 32 + hi * 16;
        v8h blo = *(const v8h*)&W1L[br];
        v8h bhi = *(const v8h*)&W1L[br + 8];
        v16h b = __builtin_shufflevector(blo, bhi, 0,1,2,3,4,5,6,7,8,9,10,11,12,13,14,15);
        acc1[n] = __builtin_amdgcn_wmma_f32_16x16x32_f16(
            false, a, false, b, (short)0, acc1[n], false, false);
      }
    }

    // bias + relu + f16, back into per-wave panel (h1: 16x256)
#pragma unroll
    for (int n = 0; n < 16; ++n) {
      float bv = b1[n * 16 + l15];
#pragma unroll
      for (int r = 0; r < 8; ++r) {
        float v = acc1[n][r] + bv;
        v = v > 0.f ? v : 0.f;
        A[(r + hi * 8) * AS + n * 16 + l15] = (half_t)v;
      }
    }

    // ---------- layer 2: [16x256] @ [256x128] ----------
    v8f acc2[8];
#pragma unroll
    for (int n = 0; n < 8; ++n) acc2[n] = (v8f){};

#pragma unroll
    for (int kc = 0; kc < 8; ++kc) {
      int ar = l15 * AS + kc * 32 + hi * 16;
      v8h alo = *(const v8h*)&A[ar];
      v8h ahi = *(const v8h*)&A[ar + 8];
      v16h a = __builtin_shufflevector(alo, ahi, 0,1,2,3,4,5,6,7,8,9,10,11,12,13,14,15);
#pragma unroll
      for (int n = 0; n < 8; ++n) {
        int br = (n * 16 + l15) * AS + kc * 32 + hi * 16;
        v8h blo = *(const v8h*)&W2L[br];
        v8h bhi = *(const v8h*)&W2L[br + 8];
        v16h b = __builtin_shufflevector(blo, bhi, 0,1,2,3,4,5,6,7,8,9,10,11,12,13,14,15);
        acc2[n] = __builtin_amdgcn_wmma_f32_16x16x32_f16(
            false, a, false, b, (short)0, acc2[n], false, false);
      }
    }

    // bias + relu + f16 -> h2 rows (16x128) in panel
#pragma unroll
    for (int n = 0; n < 8; ++n) {
      float bv = b2[n * 16 + l15];
#pragma unroll
      for (int r = 0; r < 8; ++r) {
        float v = acc2[n][r] + bv;
        v = v > 0.f ? v : 0.f;
        A[(r + hi * 8) * AS + n * 16 + l15] = (half_t)v;
      }
    }

    // ---------- layer 3: [16x128] @ [128x1] + sigmoid ----------
    float s = 0.f;
    for (int k = 0; k < 64; ++k) {
      int kk = hi * 64 + k;
      s += (float)A[l15 * AS + kk] * W3[kk];
    }
    s += __shfl_xor(s, 16, 32);               // combine half-dot products (wave32)
    if (lane < 16) {
      int e = ebase + l15;
      if (e < E) {
        float x = s + b3[0];
        out[e] = 1.f / (1.f + expf(-x));
      }
    }
  }
}

extern "C" void kernel_launch(void* const* d_in, const int* in_sizes, int n_in,
                              void* d_out, int out_size, void* d_ws, size_t ws_size,
                              hipStream_t stream) {
  const float*     z  = (const float*)d_in[0];
  const long long* ei = (const long long*)d_in[1];
  const float*     W1 = (const float*)d_in[2];
  const float*     b1 = (const float*)d_in[3];
  const float*     W2 = (const float*)d_in[4];
  const float*     b2 = (const float*)d_in[5];
  const float*     W3 = (const float*)d_in[6];
  const float*     b3 = (const float*)d_in[7];
  float* out = (float*)d_out;
  int E = in_sizes[1] / 2;

  half_t* W1T = (half_t*)d_ws;                // 256*256 f16
  half_t* W2T = W1T + 256 * 256;              // 128*256 f16

  decoder_prep<<<256, 256, 0, stream>>>(W1, W2, W1T, W2T);

  (void)hipFuncSetAttribute((const void*)decoder_mlp,
                            hipFuncAttributeMaxDynamicSharedMemorySize, LDS_BYTES);

  int nT = (E + 15) / 16;
  int blocks = (nT + WPB - 1) / WPB;
  if (blocks > 1024) blocks = 1024;           // persistent: ~few blocks per WGP max
  decoder_mlp<<<blocks, BLOCK, LDS_BYTES, stream>>>(
      z, ei, W1T, b1, W2T, b2, W3, b3, out, E);
}
